// LSHGaussian_74371653697731
// MI455X (gfx1250) — matrix operand order repaired and checked by reading, not compiled, not run on hardware
//
#include <hip/hip_runtime.h>
#include <hip/hip_bf16.h>

// Problem constants (match reference: N=16384 points, C=64 channels, D_REF=5)
#define NPTS 16384
#define CCH  64
#define DREF 5

typedef __attribute__((ext_vector_type(16))) _Float16 v16h;
typedef __attribute__((ext_vector_type(8)))  float    v8f;
typedef __attribute__((ext_vector_type(2)))  float    v2f;

#if __has_builtin(__builtin_amdgcn_wmma_f32_16x16x4_f32)
#define HAVE_GRAM_WMMA 1
#else
#define HAVE_GRAM_WMMA 0
#endif

__device__ __forceinline__ float fast_exp2(float x) {
#if __has_builtin(__builtin_amdgcn_exp2f)
  return __builtin_amdgcn_exp2f(x);   // v_exp_f32 (base-2)
#else
  return exp2f(x);
#endif
}

// pack two f32 -> f16x2 with the VOP3 CLAMP destination modifier.
// Since exp2 is monotone and >= 0:  clamp01(exp2(t)) == exp2(min(t,0)),
// which implements the reference's max(d2,0) guard for free.
__device__ __forceinline__ int cvt_pk_clamp01(float a, float b) {
  int r;
  asm("v_cvt_pk_rtz_f16_f32 %0, %1, %2 clamp" : "=v"(r) : "v"(a), "v"(b));
  return r;
}

// ---------------------------------------------------------------------------
// Prep 1: build augmented 8-dim feature vectors so that
//   refA[j] . refB[i] = log2e * (r_i.r_j - 0.5|r_i|^2 - 0.5|r_j|^2)
//                     = log2(K[i,j])   (before the >=0 clamp)
// refA[j] = (s*r0..s*r4, hs_j, 1, 0)      (j / A-side of gram WMMA)
// refB[i] = (s*r0..s*r4, 1, hs_i, 0)      (i / B-side of gram WMMA)
// with s = sqrt(log2 e), hs = -0.5*log2e*|r|^2
// ---------------------------------------------------------------------------
__global__ void prep_ref_kernel(const float* __restrict__ ref,
                                float* __restrict__ refA,
                                float* __restrict__ refB) {
  int j = blockIdx.x * blockDim.x + threadIdx.x;
  if (j >= NPTS) return;
  const float L2E = 1.4426950408889634f;   // log2(e)
  const float S   = 1.2011224087864498f;   // sqrt(log2 e)
  float r[DREF]; float sq = 0.f;
#pragma unroll
  for (int d = 0; d < DREF; ++d) { r[d] = ref[j * DREF + d]; sq += r[d] * r[d]; }
  float hs = -0.5f * L2E * sq;
  float* a = refA + (size_t)j * 8;
  float* b = refB + (size_t)j * 8;
#pragma unroll
  for (int d = 0; d < DREF; ++d) { a[d] = S * r[d]; b[d] = S * r[d]; }
  a[5] = hs;  a[6] = 1.f; a[7] = 0.f;
  b[5] = 1.f; b[6] = hs;  b[7] = 0.f;
}

// ---------------------------------------------------------------------------
// Prep 2: Uht[c][j] = (f16) U[j][c]  -- transposed f16 copy (2 MB, L2-resident)
// ---------------------------------------------------------------------------
__global__ void prep_u_kernel(const float* __restrict__ U,
                              _Float16* __restrict__ Uht) {
  int idx = blockIdx.x * blockDim.x + threadIdx.x;   // coalesced read of U
  if (idx >= NPTS * CCH) return;
  int j = idx / CCH;
  int c = idx - j * CCH;
  Uht[(size_t)c * NPTS + j] = (_Float16)U[idx];
}

// ---------------------------------------------------------------------------
// Main fused kernel. One wave owns a 16-row i-tile and all 64 channels
// (accumulated transposed: D[m=c][n=i]). 4 waves/block, no LDS tiling needed.
// ---------------------------------------------------------------------------
__global__ __launch_bounds__(128)
void lsh_gauss_main(const float* __restrict__ U,
                    const float* __restrict__ refA,
                    const float* __restrict__ refB,
                    const _Float16* __restrict__ Uht,
                    float* __restrict__ out) {
  const int lane = threadIdx.x & 31;
  const int wave = threadIdx.x >> 5;
  const int l16  = lane & 15;
  const bool hi  = (lane & 16) != 0;           // upper half-wave
  const int i0   = blockIdx.x * 64 + wave * 16;
  const int i    = i0 + l16;

#if HAVE_GRAM_WMMA
  // Loop-invariant B fragments (i-side) of the 16x16x4 f32 gram WMMA.
  // B 4x16 layout: VGPR0 = K0 (lanes<16) / K2 (lanes>=16); VGPR1 = K1 / K3.
  const float* rb = refB + (size_t)i * 8 + (hi ? 2 : 0);
  v2f bi0 = { rb[0], rb[1] };                  // K 0..3 half
  v2f bi1 = { rb[4], rb[5] };                  // K 4..7 half
#else
  float bB[8];
#pragma unroll
  for (int d = 0; d < 8; ++d) bB[d] = refB[(size_t)i * 8 + d];
#endif

  v8f acc0 = {}, acc1 = {}, acc2 = {}, acc3 = {};

  const int koffA = hi ? 16 : 0;               // K offset of U-side A fragment
  const _Float16* ubase = Uht + (size_t)l16 * NPTS;   // row c0+l16 per tile

#pragma unroll 2
  for (int j0 = 0; j0 < NPTS; j0 += 32) {
    // ---------- score tiles t0 (j0..j0+15) and t1 (j0+16..j0+31) -----------
    // D[m=j][n=i] = log2(K[i,j])  (gram of augmented features)
    v8f t0 = {}, t1 = {};
#if HAVE_GRAM_WMMA
    // A 16x4 f32 layout: VGPR0 = K0 (lanes<16) / K2 (lanes>=16); VGPR1 = K1/K3
    const float* ra0 = refA + (size_t)(j0 + l16) * 8 + (hi ? 2 : 0);
    const float* ra1 = ra0 + 16 * 8;
    v2f a00 = { ra0[0], ra0[1] };
    v2f a01 = { ra0[4], ra0[5] };
    v2f a10 = { ra1[0], ra1[1] };
    v2f a11 = { ra1[4], ra1[5] };
    t0 = __builtin_amdgcn_wmma_f32_16x16x4_f32(false, a00, false, bi0, (short)0, t0, false, false);
    t0 = __builtin_amdgcn_wmma_f32_16x16x4_f32(false, a01, false, bi1, (short)0, t0, false, false);
    t1 = __builtin_amdgcn_wmma_f32_16x16x4_f32(false, a10, false, bi0, (short)0, t1, false, false);
    t1 = __builtin_amdgcn_wmma_f32_16x16x4_f32(false, a11, false, bi1, (short)0, t1, false, false);
#else
#pragma unroll
    for (int r = 0; r < 8; ++r) {              // VALU fallback
      const float* ra = refA + (size_t)(j0 + (hi ? 8 : 0) + r) * 8;
      const float* rc = ra + 16 * 8;
      float s0 = 0.f, s1 = 0.f;
#pragma unroll
      for (int d = 0; d < 8; ++d) { s0 += ra[d] * bB[d]; s1 += rc[d] * bB[d]; }
      t0[r] = s0; t1[r] = s1;
    }
#endif

    // ---------- exp2 + clamped pack into f16 B-fragment of main WMMA -------
    // Gram D: lane n = i, VGPR r = j (r / 8+r per half-wave).  cvt_pkrtz of
    // VGPR pairs already yields B-layout (K=2r,2r+1 per dword); only the
    // half-wave ownership must be swapped via ds_swizzle SWAPX16.
    int pk0[4], pk1[4];
#pragma unroll
    for (int r = 0; r < 4; ++r) {
      float e0a = fast_exp2(t0[2 * r]);
      float e0b = fast_exp2(t0[2 * r + 1]);
      float e1a = fast_exp2(t1[2 * r]);
      float e1b = fast_exp2(t1[2 * r + 1]);
      pk0[r] = cvt_pk_clamp01(e0a, e0b);   // clamp01 == exp2(min(arg,0))
      pk1[r] = cvt_pk_clamp01(e1a, e1b);
    }
    union { v16h h; int w[8]; } bk;
#pragma unroll
    for (int r = 0; r < 4; ++r) {
      int sw0 = __builtin_amdgcn_ds_swizzle(pk0[r], 0x7C10);  // SWAPX16
      int sw1 = __builtin_amdgcn_ds_swizzle(pk1[r], 0x7C10);
      bk.w[r]     = hi ? sw1    : pk0[r];      // K = 0..15 | 16..31
      bk.w[4 + r] = hi ? pk1[r] : sw0;         // K = 8..15 swapped halves
    }

    // speculative prefetch of next chunk's j-side features
    __builtin_prefetch(refA + (size_t)(j0 + 32 + l16) * 8, 0, 1);

    // ---------- main WMMA: outT[c,i] += Uht[c, j0..j0+32) x K[j,i] ----------
#pragma unroll
    for (int t = 0; t < 4; ++t) {
      union { v16h h; uint4 u[2]; } au;
      const _Float16* up = ubase + (size_t)(t * 16) * NPTS + j0 + koffA;
      au.u[0] = *reinterpret_cast<const uint4*>(up);        // K 0-7  / 16-23
      au.u[1] = *reinterpret_cast<const uint4*>(up + 8);    // K 8-15 / 24-31
      v8f& acc = (t == 0) ? acc0 : (t == 1) ? acc1 : (t == 2) ? acc2 : acc3;
      acc = __builtin_amdgcn_wmma_f32_16x16x32_f16(false, au.h, false, bk.h,
                                                   (short)0, acc, false, false);
    }
  }

  // ---------- epilogue: out[i][c] = acc - U[i][c] ----------
  const float* urow = U   + (size_t)i * CCH;
  float*       orow = out + (size_t)i * CCH;
#pragma unroll
  for (int t = 0; t < 4; ++t) {
    const v8f& acc = (t == 0) ? acc0 : (t == 1) ? acc1 : (t == 2) ? acc2 : acc3;
    int c0 = t * 16 + (hi ? 8 : 0);            // this lane's 8 channels
    float4 u0 = *reinterpret_cast<const float4*>(urow + c0);
    float4 u1 = *reinterpret_cast<const float4*>(urow + c0 + 4);
    float4 o0 = make_float4(acc[0] - u0.x, acc[1] - u0.y,
                            acc[2] - u0.z, acc[3] - u0.w);
    float4 o1 = make_float4(acc[4] - u1.x, acc[5] - u1.y,
                            acc[6] - u1.z, acc[7] - u1.w);
    *reinterpret_cast<float4*>(orow + c0)     = o0;
    *reinterpret_cast<float4*>(orow + c0 + 4) = o1;
  }
}

// ---------------------------------------------------------------------------
extern "C" void kernel_launch(void* const* d_in, const int* in_sizes, int n_in,
                              void* d_out, int out_size, void* d_ws, size_t ws_size,
                              hipStream_t stream) {
  const float* U   = (const float*)d_in[0];   // [N, C] f32
  const float* ref = (const float*)d_in[1];   // [N, 5] f32
  float* out = (float*)d_out;                 // [N, C] f32

  // workspace layout: refA (512KB) | refB (512KB) | Uht f16 (2MB)
  char* ws = (char*)d_ws;
  float*    refA = (float*)ws;
  float*    refB = (float*)(ws + (size_t)NPTS * 8 * sizeof(float));
  _Float16* Uht  = (_Float16*)(ws + 2 * (size_t)NPTS * 8 * sizeof(float));

  prep_ref_kernel<<<NPTS / 256, 256, 0, stream>>>(ref, refA, refB);
  prep_u_kernel<<<(NPTS * CCH) / 256, 256, 0, stream>>>(U, Uht);
  // 256 blocks x 4 waves; each wave: 16 rows x 64 channels, 512 j-chunks
  lsh_gauss_main<<<NPTS / 64, 128, 0, stream>>>(U, refA, refB, Uht, out);
}
